// FLAMamba_7756710936887
// MI455X (gfx1250) — matrix operand order repaired and checked by reading, not compiled, not run on hardware
//
#include <hip/hip_runtime.h>
#include <hip/hip_bf16.h>
#include <cstdint>
#include <cstddef>

// Problem dims (fixed by reference)
#define B_   2
#define L_   2048
#define DM_  1024
#define DI_  2048
#define DS_  16
#define DR_  64
#define ML_  (B_ * L_)        // 4096 rows of "tokens"
#define XZW_ (2 * DI_)        // 4096 (xz width)
#define XDW_ (DR_ + 2 * DS_)  // 96   (x_dbl width)
#define NCH_ 32               // scan chunks
#define TC_  (L_ / NCH_)      // 64 steps per chunk

typedef _Float16 half8  __attribute__((ext_vector_type(8)));
typedef _Float16 half16 __attribute__((ext_vector_type(16)));
typedef float    float8 __attribute__((ext_vector_type(8)));

__device__ __forceinline__ float sigmoidf_(float x) { return 1.f / (1.f + __expf(-x)); }
__device__ __forceinline__ float softplusf_(float x) {
  // log(1+exp(x)), stable both tails
  return fmaxf(x, 0.f) + log1pf(__expf(-fabsf(x)));
}

__device__ __forceinline__ half16 ld_afrag(const _Float16* row, int kk, int kh) {
  half8 lo = *(const half8*)(row + kk + kh * 16);
  half8 hi = *(const half8*)(row + kk + kh * 16 + 8);
  half16 a;
#pragma unroll
  for (int i = 0; i < 8; ++i) { a[i] = lo[i]; a[8 + i] = hi[i]; }
  return a;
}

// ---------------------------------------------------------------------------
// f32 -> f16 conversion (grid-stride)
// ---------------------------------------------------------------------------
__global__ void cvt_f32_f16(const float* __restrict__ in, _Float16* __restrict__ out, int n) {
  int i = blockIdx.x * blockDim.x + threadIdx.x;
  int stride = gridDim.x * blockDim.x;
  for (; i < n; i += stride) out[i] = (_Float16)in[i];
}

// ---------------------------------------------------------------------------
// WMMA GEMM (16x16 per wave): C[M,N](f32) = A[M,K](f16) * W[N,K]^T (f16)
// Used for the skinny x_proj GEMM (N = 96).
// Fragment layouts per CDNA5 ISA 7.12.2 (wave32):
//   A 16x32 f16: lane m=l&15, K sub-block by l>>4
//   B 32x16 f16: lane n=l&15; code's own k order (same as A): lanes 0-15 K=0..15, lanes 16-31 K=16..31
//   C 16x16 f32: elem v -> M = v + 8*(l>>4), N = l&15
// ---------------------------------------------------------------------------
__global__ __launch_bounds__(256)
void gemm_f16_nt(const _Float16* __restrict__ A, int lda,
                 const _Float16* __restrict__ W, int ldw,
                 float* __restrict__ C, int ldc, int K) {
  const int lane = threadIdx.x & 31;
  const int wave = threadIdx.x >> 5;
  const int m0 = blockIdx.y * 64 + (wave >> 1) * 16;
  const int n0 = blockIdx.x * 32 + (wave & 1) * 16;
  const int kh = lane >> 4;
  const int r  = lane & 15;

  const _Float16* Abase = A + (size_t)(m0 + r) * lda;
  const _Float16* Wbase = W + (size_t)(n0 + r) * ldw + kh * 16;

  float8 acc = {0.f, 0.f, 0.f, 0.f, 0.f, 0.f, 0.f, 0.f};
  for (int kk = 0; kk < K; kk += 32) {
    half16 a = ld_afrag(Abase, kk, kh);
    half16 b = *(const half16*)(Wbase + kk);
    acc = __builtin_amdgcn_wmma_f32_16x16x32_f16(false, a, false, b,
                                                 (short)0, acc, false, false);
  }

  const int n  = n0 + r;
  const int mb = m0 + kh * 8;
#pragma unroll
  for (int v = 0; v < 8; ++v) C[(size_t)(mb + v) * ldc + n] = acc[v];
}

// ---------------------------------------------------------------------------
// WMMA GEMM (32x32 per wave, 2x2 fragments): doubles arithmetic intensity by
// reusing each A/B fragment across two WMMAs. Block = 8 waves (2 M x 4 N)
// -> 64(M) x 128(N) block tile. Requires M % 64 == 0, N % 128 == 0, K % 32 == 0.
// ---------------------------------------------------------------------------
__global__ __launch_bounds__(256)
void gemm_f16_nt_2x2(const _Float16* __restrict__ A, int lda,
                     const _Float16* __restrict__ W, int ldw,
                     float* __restrict__ C, int ldc, int K) {
  const int lane = threadIdx.x & 31;
  const int wave = threadIdx.x >> 5;
  const int m0 = blockIdx.y * 64 + (wave >> 2) * 32;   // wave tile 32 in M
  const int n0 = blockIdx.x * 128 + (wave & 3) * 32;   // wave tile 32 in N
  const int kh = lane >> 4;
  const int r  = lane & 15;

  const _Float16* A0 = A + (size_t)(m0 + r) * lda;
  const _Float16* A1 = A0 + (size_t)16 * lda;
  const _Float16* W0 = W + (size_t)(n0 + r) * ldw + kh * 16;
  const _Float16* W1 = W0 + (size_t)16 * ldw;

  float8 acc00 = {0.f, 0.f, 0.f, 0.f, 0.f, 0.f, 0.f, 0.f};
  float8 acc01 = acc00, acc10 = acc00, acc11 = acc00;

  for (int kk = 0; kk < K; kk += 32) {
    // prefetch A 8 k-tiles (256 elems) ahead into caches while WMMAs retire (global_prefetch_b8)
    if (kk + 256 < K) __builtin_prefetch(A0 + kk + 256, 0, 3);

    half16 a0 = ld_afrag(A0, kk, kh);
    half16 a1 = ld_afrag(A1, kk, kh);
    half16 b0 = *(const half16*)(W0 + kk);
    half16 b1 = *(const half16*)(W1 + kk);

    acc00 = __builtin_amdgcn_wmma_f32_16x16x32_f16(false, a0, false, b0,
                                                   (short)0, acc00, false, false);
    acc01 = __builtin_amdgcn_wmma_f32_16x16x32_f16(false, a0, false, b1,
                                                   (short)0, acc01, false, false);
    acc10 = __builtin_amdgcn_wmma_f32_16x16x32_f16(false, a1, false, b0,
                                                   (short)0, acc10, false, false);
    acc11 = __builtin_amdgcn_wmma_f32_16x16x32_f16(false, a1, false, b1,
                                                   (short)0, acc11, false, false);
  }

  const int n_lo = n0 + r;
  const int n_hi = n0 + 16 + r;
  const int mb_lo = m0 + kh * 8;
  const int mb_hi = m0 + 16 + kh * 8;
#pragma unroll
  for (int v = 0; v < 8; ++v) {
    C[(size_t)(mb_lo + v) * ldc + n_lo] = acc00[v];
    C[(size_t)(mb_lo + v) * ldc + n_hi] = acc01[v];
    C[(size_t)(mb_hi + v) * ldc + n_lo] = acc10[v];
    C[(size_t)(mb_hi + v) * ldc + n_hi] = acc11[v];
  }
}

// ---------------------------------------------------------------------------
// Causal depthwise conv (DC=4) + bias + SiLU; reads xs = xz[:, 0:DI].
// Writes u as f32 (scan) and f16 (next GEMM).
// ---------------------------------------------------------------------------
__global__ void conv_silu_k(const float* __restrict__ xz, const float* __restrict__ cw,
                            const float* __restrict__ cb, float* __restrict__ u32,
                            _Float16* __restrict__ u16) {
  const int total = ML_ * DI_;
  int idx = blockIdx.x * blockDim.x + threadIdx.x;
  int stride = gridDim.x * blockDim.x;
  for (; idx < total; idx += stride) {
    int d   = idx % DI_;
    int row = idx / DI_;      // b*L + l
    int l   = row % L_;
    float acc = cb[d];
#pragma unroll
    for (int j = 0; j < 4; ++j) {
      int t = l - 3 + j;
      if (t >= 0) acc += xz[(size_t)(row - 3 + j) * XZW_ + d] * cw[d * 4 + j];
    }
    float v = acc * sigmoidf_(acc);
    u32[idx] = v;
    u16[idx] = (_Float16)v;
  }
}

// ---------------------------------------------------------------------------
// Selective scan, 3-pass chunked prefix over L.
// h_t[s] = exp(dt_t*A[s]) * h_{t-1}[s] + (dt_t*u_t) * B_t[s]
// ---------------------------------------------------------------------------
__global__ void scan_pass1(const float* __restrict__ dtlin, const float* __restrict__ dtb,
                           const float* __restrict__ Alog, const float* __restrict__ u32,
                           const float* __restrict__ xdbl,
                           float* __restrict__ hend, float* __restrict__ aprod) {
  int idx = blockIdx.x * blockDim.x + threadIdx.x;   // over B*DI*NCH
  if (idx >= B_ * DI_ * NCH_) return;
  int chunk = idx % NCH_;
  int d     = (idx / NCH_) % DI_;
  int b     = idx / (NCH_ * DI_);

  float Aneg[DS_];
#pragma unroll
  for (int s = 0; s < DS_; ++s) Aneg[s] = -__expf(Alog[d * DS_ + s]);
  const float bias = dtb[d];

  float h[DS_], ap[DS_];
#pragma unroll
  for (int s = 0; s < DS_; ++s) { h[s] = 0.f; ap[s] = 1.f; }

  const int t0 = chunk * TC_;
  for (int t = t0; t < t0 + TC_; ++t) {
    size_t row = (size_t)b * L_ + t;
    float dtv = softplusf_(dtlin[row * DI_ + d] + bias);
    float du  = dtv * u32[row * DI_ + d];
    const float* xd = xdbl + row * XDW_;
#pragma unroll
    for (int s = 0; s < DS_; ++s) {
      float da = __expf(dtv * Aneg[s]);
      h[s]  = h[s] * da + du * xd[DR_ + s];
      ap[s] *= da;
    }
  }
  size_t base = ((size_t)(b * DI_ + d) * NCH_ + chunk) * DS_;
#pragma unroll
  for (int s = 0; s < DS_; ++s) { hend[base + s] = h[s]; aprod[base + s] = ap[s]; }
}

__global__ void scan_pass2(const float* __restrict__ hend, const float* __restrict__ aprod,
                           float* __restrict__ hin) {
  int idx = blockIdx.x * blockDim.x + threadIdx.x;   // over B*DI*DS
  if (idx >= B_ * DI_ * DS_) return;
  int s = idx % DS_;
  int d = (idx / DS_) % DI_;
  int b = idx / (DS_ * DI_);
  size_t cbase = (size_t)(b * DI_ + d) * NCH_;
  float h = 0.f;
  for (int c = 0; c < NCH_; ++c) {
    size_t o = (cbase + c) * DS_ + s;
    hin[o] = h;                        // state entering chunk c
    h = hend[o] + aprod[o] * h;
  }
}

__global__ void scan_pass3(const float* __restrict__ dtlin, const float* __restrict__ dtb,
                           const float* __restrict__ Alog, const float* __restrict__ u32,
                           const float* __restrict__ xdbl, const float* __restrict__ hin,
                           const float* __restrict__ xz, const float* __restrict__ Dp,
                           _Float16* __restrict__ y16) {
  int idx = blockIdx.x * blockDim.x + threadIdx.x;   // over B*DI*NCH
  if (idx >= B_ * DI_ * NCH_) return;
  int chunk = idx % NCH_;
  int d     = (idx / NCH_) % DI_;
  int b     = idx / (NCH_ * DI_);

  float Aneg[DS_];
#pragma unroll
  for (int s = 0; s < DS_; ++s) Aneg[s] = -__expf(Alog[d * DS_ + s]);
  const float bias = dtb[d];
  const float Dv   = Dp[d];

  float h[DS_];
  {
    size_t base = ((size_t)(b * DI_ + d) * NCH_ + chunk) * DS_;
#pragma unroll
    for (int s = 0; s < DS_; ++s) h[s] = hin[base + s];
  }

  const int t0 = chunk * TC_;
  for (int t = t0; t < t0 + TC_; ++t) {
    size_t row = (size_t)b * L_ + t;
    float uu  = u32[row * DI_ + d];
    float dtv = softplusf_(dtlin[row * DI_ + d] + bias);
    float du  = dtv * uu;
    const float* xd = xdbl + row * XDW_;
    float y = 0.f;
#pragma unroll
    for (int s = 0; s < DS_; ++s) {
      float da = __expf(dtv * Aneg[s]);
      h[s] = h[s] * da + du * xd[DR_ + s];
      y += h[s] * xd[DR_ + DS_ + s];
    }
    float z = xz[row * XZW_ + DI_ + d];
    float o = (y + uu * Dv) * (z * sigmoidf_(z));
    y16[row * DI_ + d] = (_Float16)o;
  }
}

// ---------------------------------------------------------------------------
// Launch
// ---------------------------------------------------------------------------
extern "C" void kernel_launch(void* const* d_in, const int* in_sizes, int n_in,
                              void* d_out, int out_size, void* d_ws, size_t ws_size,
                              hipStream_t stream) {
  const float* x     = (const float*)d_in[0];   // [B,L,DM]
  const float* winp  = (const float*)d_in[1];   // [2*DI, DM]
  const float* convw = (const float*)d_in[2];   // [DI,1,4]
  const float* convb = (const float*)d_in[3];   // [DI]
  const float* wxp   = (const float*)d_in[4];   // [96, DI]
  const float* wdt   = (const float*)d_in[5];   // [DI, 64]
  const float* dtb   = (const float*)d_in[6];   // [DI]
  const float* Alog  = (const float*)d_in[7];   // [DI, DS]
  const float* Dp    = (const float*)d_in[8];   // [DI]
  const float* wout  = (const float*)d_in[9];   // [DM, DI]
  float* out = (float*)d_out;                   // [B,L,DM]

  char* w = (char*)d_ws;
  size_t off = 0;
  auto alloc = [&](size_t bytes) -> void* {
    void* p = w + off;
    off = (off + bytes + 255) & ~(size_t)255;
    return p;
  };

  float*    xz     = (float*)   alloc((size_t)ML_ * XZW_ * 4);  // 64 MB
  _Float16* x16    = (_Float16*)alloc((size_t)ML_ * DM_ * 2);   //  8 MB
  _Float16* winp16 = (_Float16*)alloc((size_t)XZW_ * DM_ * 2);  //  8 MB
  float*    u32    = (float*)   alloc((size_t)ML_ * DI_ * 4);   // 32 MB
  _Float16* u16    = (_Float16*)alloc((size_t)ML_ * DI_ * 2);   // 16 MB
  float*    xdbl   = (float*)   alloc((size_t)ML_ * XDW_ * 4);
  _Float16* xdbl16 = (_Float16*)alloc((size_t)ML_ * XDW_ * 2);
  _Float16* wxp16  = (_Float16*)alloc((size_t)XDW_ * DI_ * 2);
  _Float16* wdt16  = (_Float16*)alloc((size_t)DI_ * DR_ * 2);
  _Float16* wout16 = (_Float16*)alloc((size_t)DM_ * DI_ * 2);
  float*    dtlin  = (float*)   alloc((size_t)ML_ * DI_ * 4);   // 32 MB
  float*    hend   = (float*)   alloc((size_t)B_ * DI_ * NCH_ * DS_ * 4);
  float*    aprod  = (float*)   alloc((size_t)B_ * DI_ * NCH_ * DS_ * 4);
  float*    hin    = (float*)   alloc((size_t)B_ * DI_ * NCH_ * DS_ * 4);
  _Float16* y16    = (_Float16*)alloc((size_t)ML_ * DI_ * 2);

  const int TB = 256;
  auto cvt = [&](const float* src, _Float16* dst, int n) {
    int blocks = (n + TB - 1) / TB;
    if (blocks > 4096) blocks = 4096;
    cvt_f32_f16<<<blocks, TB, 0, stream>>>(src, dst, n);
  };

  // 0) f16 conversions of GEMM operands
  cvt(x,    x16,    ML_ * DM_);
  cvt(winp, winp16, XZW_ * DM_);
  cvt(wxp,  wxp16,  XDW_ * DI_);
  cvt(wdt,  wdt16,  DI_ * DR_);
  cvt(wout, wout16, DM_ * DI_);

  // 1) xz = x @ in_proj_w^T   [4096 x 4096, K=1024]
  gemm_f16_nt_2x2<<<dim3(XZW_ / 128, ML_ / 64), TB, 0, stream>>>(
      x16, DM_, winp16, DM_, xz, XZW_, DM_);

  // 2) u = silu(conv(xs) + b)
  conv_silu_k<<<8192, TB, 0, stream>>>(xz, convw, convb, u32, u16);

  // 3) x_dbl = u @ x_proj_w^T   [4096 x 96, K=2048]  (skinny N -> 16x16 kernel)
  gemm_f16_nt<<<dim3(XDW_ / 32, ML_ / 64), TB, 0, stream>>>(
      u16, DI_, wxp16, DI_, xdbl, XDW_, DI_);
  cvt(xdbl, xdbl16, ML_ * XDW_);

  // 4) dtlin = x_dbl[:, :64] @ dt_proj_w^T   [4096 x 2048, K=64]
  gemm_f16_nt_2x2<<<dim3(DI_ / 128, ML_ / 64), TB, 0, stream>>>(
      xdbl16, XDW_, wdt16, DR_, dtlin, DI_, DR_);

  // 5) chunked selective scan + gating epilogue -> y16
  scan_pass1<<<(B_ * DI_ * NCH_ + TB - 1) / TB, TB, 0, stream>>>(
      dtlin, dtb, Alog, u32, xdbl, hend, aprod);
  scan_pass2<<<(B_ * DI_ * DS_ + TB - 1) / TB, TB, 0, stream>>>(hend, aprod, hin);
  scan_pass3<<<(B_ * DI_ * NCH_ + TB - 1) / TB, TB, 0, stream>>>(
      dtlin, dtb, Alog, u32, xdbl, hin, xz, Dp, y16);

  // 6) out = y @ out_proj_w^T   [4096 x 1024, K=2048]
  gemm_f16_nt_2x2<<<dim3(DM_ / 128, ML_ / 64), TB, 0, stream>>>(
      y16, DI_, wout16, DI_, out, DM_, DI_);
}
